// VariationalStateSpaceModel_66597762891932
// MI455X (gfx1250) — matrix operand order repaired and checked by reading, not compiled, not run on hardware
//
#include <hip/hip_runtime.h>
#include <hip/hip_bf16.h>
#include <math.h>

// ---------------------------------------------------------------------------
// Types for CDNA5 WMMA (wave32)
// ---------------------------------------------------------------------------
typedef __attribute__((ext_vector_type(16))) __bf16 v16bf;
typedef __attribute__((ext_vector_type(8)))  __bf16 v8bf;
typedef __attribute__((ext_vector_type(8)))  float  v8f;

#define TT 512
#define BB 128
#define DD 128
#define SS 256
#define OO 32
#define RR 4
#define NSTEPS 4
#define G3 (3*SS)          // 768

__device__ __forceinline__ float sigmoidf_(float x){ return 1.0f/(1.0f+__expf(-x)); }
__device__ __forceinline__ float geluf_(float x){ return 0.5f*x*(1.0f+erff(x*0.7071067811865476f)); }
__device__ __forceinline__ float clampf_(float x, float lo, float hi){ return fminf(fmaxf(x,lo),hi); }

// ---------------------------------------------------------------------------
// Pack weight W (Nreal x K, row-major, fp32) into bf16 WMMA B-fragments.
// Fragment layout (per CDNA5 ISA, dense 16-bit B 32x16):
//   frag[((kt*Ntiles + nt)*32 + lane)*16 + i] = W[nt*16 + lane%16][kt*32 + (lane>=16?8:0) + (i<8 ? i : i+8)]
// Ntiles is PADDED to a multiple of 4; out-of-range columns are zero-filled,
// so the GEMM can always run 4 unconditional WMMAs per k-step.
// ---------------------------------------------------------------------------
__global__ void pack_bfrag_kernel(const float* __restrict__ W, __bf16* __restrict__ out,
                                  int Nreal, int K, int Ntiles)
{
    int Ktiles = K >> 5;
    size_t total = (size_t)Ktiles * Ntiles * 512;
    size_t idx = (size_t)blockIdx.x * blockDim.x + threadIdx.x;
    if (idx >= total) return;
    size_t tile = idx >> 9;
    int e    = (int)(idx & 511);
    int lane = e >> 4;
    int i    = e & 15;
    int kt = (int)(tile / Ntiles);
    int nt = (int)(tile % Ntiles);
    int k  = kt*32 + ((lane >= 16) ? 8 : 0) + ((i < 8) ? i : i + 8);
    int n  = nt*16 + (lane & 15);
    float v = (n < Nreal) ? W[(size_t)n*K + k] : 0.0f;
    out[idx] = (__bf16)v;
}

// ---------------------------------------------------------------------------
// Generic WMMA GEMM:  C[M x N] = act( A_f32[M x K] * Bfrag + bias )
// Register-blocked 1 M-tile x 4 N-tiles per wave, fully straight-line:
// NtilesPad is a multiple of 4 (fragments zero-padded), so the inner loop is
// 1 A load+convert and 4 unconditional B-load+WMMA pairs. EXEC stays all-ones
// through every WMMA. Per-lane n<Nvalid guard applies only at the store.
// A-fragment per ISA: lane l: m=l%16, kb=(l<16?0:8):
//   a[0..7]  = A[m][kt*32 + kb      .. +7]
//   a[8..15] = A[m][kt*32 + 16 + kb .. +7]
// ---------------------------------------------------------------------------
__global__ __launch_bounds__(256) void gemm_bf16_wmma(
    const float* __restrict__ A, long lda,
    const __bf16* __restrict__ Bfrag, const float* __restrict__ bias,
    float* __restrict__ C, long ldc,
    int Mtiles, int NtilesPad, int Ktiles, int Nvalid, int act)
{
    int Ngroups = NtilesPad >> 2;
    long wave = (long)blockIdx.x * (blockDim.x >> 5) + (threadIdx.x >> 5);
    if (wave >= (long)Mtiles * Ngroups) return;
    int mt  = (int)(wave / Ngroups);
    int ng  = (int)(wave % Ngroups);
    int nt0 = ng * 4;

    int lane = threadIdx.x & 31;
    int ml   = lane & 15;
    int hih  = lane >> 4;

    const float* arow = A + (size_t)(mt*16 + ml) * lda + hih*8;
    v8f acc0 = {0.f,0.f,0.f,0.f,0.f,0.f,0.f,0.f};
    v8f acc1 = acc0, acc2 = acc0, acc3 = acc0;

    for (int kt = 0; kt < Ktiles; ++kt) {
        const float* ap = arow + (size_t)kt*32;
        float4 f0 = *(const float4*)(ap + 0);
        float4 f1 = *(const float4*)(ap + 4);
        float4 f2 = *(const float4*)(ap + 16);
        float4 f3 = *(const float4*)(ap + 20);
        v16bf a;
        a[0]=(__bf16)f0.x; a[1]=(__bf16)f0.y; a[2]=(__bf16)f0.z; a[3]=(__bf16)f0.w;
        a[4]=(__bf16)f1.x; a[5]=(__bf16)f1.y; a[6]=(__bf16)f1.z; a[7]=(__bf16)f1.w;
        a[8]=(__bf16)f2.x; a[9]=(__bf16)f2.y; a[10]=(__bf16)f2.z; a[11]=(__bf16)f2.w;
        a[12]=(__bf16)f3.x; a[13]=(__bf16)f3.y; a[14]=(__bf16)f3.z; a[15]=(__bf16)f3.w;

        const __bf16* fb = Bfrag + ((size_t)kt*NtilesPad + nt0)*512 + (size_t)lane*16;
        v16bf b0 = *(const v16bf*)(fb);
        v16bf b1 = *(const v16bf*)(fb + 512);
        v16bf b2 = *(const v16bf*)(fb + 1024);
        v16bf b3 = *(const v16bf*)(fb + 1536);

        acc0 = __builtin_amdgcn_wmma_f32_16x16x32_bf16(false, a, false, b0, (short)0, acc0, false, false);
        acc1 = __builtin_amdgcn_wmma_f32_16x16x32_bf16(false, a, false, b1, (short)0, acc1, false, false);
        acc2 = __builtin_amdgcn_wmma_f32_16x16x32_bf16(false, a, false, b2, (short)0, acc2, false, false);
        acc3 = __builtin_amdgcn_wmma_f32_16x16x32_bf16(false, a, false, b3, (short)0, acc3, false, false);
    }

    const size_t mrow = (size_t)(mt*16 + hih*8);
    {
        int n = (nt0 + 0)*16 + ml;
        if (n < Nvalid) {
            float bv = bias ? bias[n] : 0.0f;
            #pragma unroll
            for (int v = 0; v < 8; ++v) {
                float r = acc0[v] + bv;
                if (act == 1) r = geluf_(r);
                C[(mrow + v) * ldc + n] = r;
            }
        }
    }
    {
        int n = (nt0 + 1)*16 + ml;
        if (n < Nvalid) {
            float bv = bias ? bias[n] : 0.0f;
            #pragma unroll
            for (int v = 0; v < 8; ++v) {
                float r = acc1[v] + bv;
                if (act == 1) r = geluf_(r);
                C[(mrow + v) * ldc + n] = r;
            }
        }
    }
    {
        int n = (nt0 + 2)*16 + ml;
        if (n < Nvalid) {
            float bv = bias ? bias[n] : 0.0f;
            #pragma unroll
            for (int v = 0; v < 8; ++v) {
                float r = acc2[v] + bv;
                if (act == 1) r = geluf_(r);
                C[(mrow + v) * ldc + n] = r;
            }
        }
    }
    {
        int n = (nt0 + 3)*16 + ml;
        if (n < Nvalid) {
            float bv = bias ? bias[n] : 0.0f;
            #pragma unroll
            for (int v = 0; v < 8; ++v) {
                float r = acc3[v] + bv;
                if (act == 1) r = geluf_(r);
                C[(mrow + v) * ldc + n] = r;
            }
        }
    }
}

// ---------------------------------------------------------------------------
// Persistent GRU scan. One workgroup (1024 thr = 32 waves) runs one scan;
// grid may be 2 (fwd + bwd encoder concurrently on two WGPs).
// h kept as bf16 in LDS, double-buffered (2*128*256*2B = 128KB dynamic LDS).
// Each step: gh = h @ Whh^T (WMMA over L2-resident bf16 fragments), then
// fused GRU gate math with precomputed gx[t], new h -> LDS + global out.
// The single A fragment feeds all three gate WMMAs per k-step.
// ---------------------------------------------------------------------------
__global__ __launch_bounds__(1024) void gru_scan_kernel(
    const float* __restrict__ gx0, const __bf16* __restrict__ whh0, const float* __restrict__ bhh0,
    float* __restrict__ out0, int col0, int ldc0, int rev0,
    const float* __restrict__ gx1, const __bf16* __restrict__ whh1, const float* __restrict__ bhh1,
    float* __restrict__ out1, int col1, int ldc1, int rev1,
    int T, int tmajor, float* __restrict__ hlast)
{
    extern __shared__ __bf16 hsm[];   // 2 * BB*SS bf16

    const float*  gx;  const __bf16* whh; const float* bhh;
    float* out; int col, ldc, rev;
    if (blockIdx.x == 0) { gx=gx0; whh=whh0; bhh=bhh0; out=out0; col=col0; ldc=ldc0; rev=rev0; }
    else                 { gx=gx1; whh=whh1; bhh=bhh1; out=out1; col=col1; ldc=ldc1; rev=rev1; }

    for (int idx = threadIdx.x; idx < BB*SS; idx += blockDim.x)
        hsm[idx] = (__bf16)0.0f;      // h0 = 0 in buffer 0
    __syncthreads();

    const int wave = threadIdx.x >> 5;
    const int lane = threadIdx.x & 31;
    const int ml   = lane & 15;
    const int hih  = lane >> 4;

    int cur = 0;
    for (int i = 0; i < T; ++i) {
        const int tt = rev ? (T - 1 - i) : i;

        // Software prefetch of next step's gx (hint only).
        if (i + 1 < T) {
            const int tn = rev ? (T - 2 - i) : (i + 1);
            const int lines = (BB*G3) / 32;   // 32 floats = 128B line
            if (tmajor) {
                const float* pfb = gx + (size_t)tn * BB * G3;
                for (int q = threadIdx.x; q < lines; q += blockDim.x)
                    __builtin_prefetch(pfb + (size_t)q*32, 0, 0);
            } else {
                for (int q = threadIdx.x; q < lines; q += blockDim.x) {
                    int m = q / (G3/32);
                    int o = (q % (G3/32)) * 32;
                    __builtin_prefetch(gx + ((size_t)m*T + tn)*G3 + o, 0, 0);
                }
            }
        }

        const __bf16* hbufc = hsm + (size_t)cur       * (BB*SS);
        __bf16*       hbufn = hsm + (size_t)(cur ^ 1) * (BB*SS);

        // 32 waves x 4 (m,s)-tile pairs = 128 pairs = all of (8 mtiles x 16 stiles)
        for (int p = 0; p < 4; ++p) {
            const int pair = wave*4 + p;
            const int mt = pair >> 4;
            const int st = pair & 15;

            v8f ar = {0.f,0.f,0.f,0.f,0.f,0.f,0.f,0.f};
            v8f az = ar, an = ar;

            const __bf16* hrow = hbufc + (size_t)(mt*16 + ml)*SS + hih*8;
            for (int kt = 0; kt < SS/32; ++kt) {
                v8bf l0 = *(const v8bf*)(hrow + kt*32);
                v8bf l1 = *(const v8bf*)(hrow + kt*32 + 16);
                v16bf a;
                #pragma unroll
                for (int j = 0; j < 8; ++j) { a[j] = l0[j]; a[8+j] = l1[j]; }

                const __bf16* fb = whh + (size_t)lane*16;
                v16bf br = *(const v16bf*)(fb + (size_t)(kt*48 +      st)*512);
                v16bf bz = *(const v16bf*)(fb + (size_t)(kt*48 + 16 + st)*512);
                v16bf bn = *(const v16bf*)(fb + (size_t)(kt*48 + 32 + st)*512);

                ar = __builtin_amdgcn_wmma_f32_16x16x32_bf16(false, a, false, br, (short)0, ar, false, false);
                az = __builtin_amdgcn_wmma_f32_16x16x32_bf16(false, a, false, bz, (short)0, az, false, false);
                an = __builtin_amdgcn_wmma_f32_16x16x32_bf16(false, a, false, bn, (short)0, an, false, false);
            }

            const int s = st*16 + ml;
            const float bhr = bhh[s], bhz = bhh[SS+s], bhn = bhh[2*SS+s];
            #pragma unroll
            for (int v = 0; v < 8; ++v) {
                const int m = mt*16 + hih*8 + v;
                const size_t gxb = tmajor ? ((size_t)tt*BB + m)*G3
                                          : ((size_t)m*T + tt)*G3;
                const float xr = gx[gxb + s];
                const float xz = gx[gxb + SS + s];
                const float xn = gx[gxb + 2*SS + s];
                const float hold = (float)hbufc[(size_t)m*SS + s];
                const float r  = sigmoidf_(xr + ar[v] + bhr);
                const float z  = sigmoidf_(xz + az[v] + bhz);
                const float nn = tanhf(xn + r*(an[v] + bhn));
                const float hnew = (1.0f - z)*nn + z*hold;
                hbufn[(size_t)m*SS + s] = (__bf16)hnew;
                out[((size_t)tt*BB + m)*ldc + col + s] = hnew;
            }
        }
        __syncthreads();
        cur ^= 1;
    }

    if (hlast && blockIdx.x == 0) {
        const __bf16* hf = hsm + (size_t)cur * (BB*SS);
        for (int idx = threadIdx.x; idx < BB*SS; idx += blockDim.x)
            hlast[idx] = (float)hf[idx];
    }
}

// ---------------------------------------------------------------------------
// z = post_mean + exp(0.5*clip(post_logvar)) * eps^T    (z is (T,B,S))
// ---------------------------------------------------------------------------
__global__ void build_z_kernel(const float* __restrict__ pp, const float* __restrict__ eps,
                               float* __restrict__ z)
{
    size_t idx = (size_t)blockIdx.x * blockDim.x + threadIdx.x;
    size_t total = (size_t)TT*BB*SS;
    if (idx >= total) return;
    int t = (int)(idx / (BB*SS));
    int r = (int)(idx % (BB*SS));
    int b = r >> 8;           // /SS
    int s = r & 255;          // %SS
    const float* row = pp + ((size_t)t*BB + b) * (2*SS);
    float qm  = row[s];
    float qlv = clampf_(row[SS + s], -10.0f, 5.0f);
    z[idx] = qm + __expf(0.5f*qlv) * eps[((size_t)b*TT + t)*SS + s];
}

// ---------------------------------------------------------------------------
// Fused GRU gate combine for the rollout (biases already folded by GEMMs).
// ---------------------------------------------------------------------------
__global__ void gru_gate_kernel(const float* __restrict__ g, const float* __restrict__ gh,
                                const float* __restrict__ hold, float* __restrict__ hnew)
{
    int idx = blockIdx.x * blockDim.x + threadIdx.x;
    if (idx >= BB*SS) return;
    int b = idx >> 8;
    int s = idx & 255;
    const float* gr  = g  + (size_t)b*G3;
    const float* ghr = gh + (size_t)b*G3;
    float r = sigmoidf_(gr[s]      + ghr[s]);
    float z = sigmoidf_(gr[SS+s]   + ghr[SS+s]);
    float n = tanhf    (gr[2*SS+s] + r*ghr[2*SS+s]);
    hnew[idx] = (1.0f - z)*n + z*hold[idx];
}

__global__ void extract_mean_kernel(const float* __restrict__ pr, float* __restrict__ st,
                                    float* __restrict__ fs, int step)
{
    int idx = blockIdx.x * blockDim.x + threadIdx.x;
    if (idx >= BB*SS) return;
    int b = idx >> 8;
    int s = idx & 255;
    float m = pr[(size_t)b*(2*SS) + s];
    st[idx] = m;
    fs[((size_t)b*NSTEPS + step)*SS + s] = m;
}

// ---------------------------------------------------------------------------
// Row LayerNorm (over 256) + exact GELU.  One block (256 thr) per row.
// ---------------------------------------------------------------------------
__global__ __launch_bounds__(256) void ln_gelu_kernel(const float* __restrict__ x,
                                                      const float* __restrict__ g,
                                                      const float* __restrict__ b,
                                                      float* __restrict__ out, int rows)
{
    __shared__ float red[256];
    int row = blockIdx.x;
    if (row >= rows) return;
    float v = x[(size_t)row*SS + threadIdx.x];
    red[threadIdx.x] = v; __syncthreads();
    for (int o = 128; o > 0; o >>= 1) { if (threadIdx.x < o) red[threadIdx.x] += red[threadIdx.x+o]; __syncthreads(); }
    float mean = red[0] * (1.0f/SS); __syncthreads();
    float d = v - mean;
    red[threadIdx.x] = d*d; __syncthreads();
    for (int o = 128; o > 0; o >>= 1) { if (threadIdx.x < o) red[threadIdx.x] += red[threadIdx.x+o]; __syncthreads(); }
    float var = red[0] * (1.0f/SS);
    float y = d * rsqrtf(var + 1e-5f) * g[threadIdx.x] + b[threadIdx.x];
    out[(size_t)row*SS + threadIdx.x] = geluf_(y);
}

// ---------------------------------------------------------------------------
// KL: partial sum per time step, then final mean.
// ---------------------------------------------------------------------------
__global__ __launch_bounds__(256) void kl_partial_kernel(const float* __restrict__ pp,
                                                         const float* __restrict__ pr,
                                                         float* __restrict__ partial)
{
    __shared__ float red[256];
    int t = blockIdx.x;
    float acc = 0.0f;
    for (int idx = threadIdx.x; idx < BB*SS; idx += 256) {
        int b = idx >> 8;
        int s = idx & 255;
        const float* qp = pp + ((size_t)t*BB + b)*(2*SS);
        float qm  = qp[s];
        float qlv = clampf_(qp[SS+s], -10.0f, 5.0f);
        float pm = 0.0f, plv = 0.0f;
        if (t > 0) {
            const float* pq = pr + ((size_t)(t-1)*BB + b)*(2*SS);
            pm  = pq[s];
            plv = clampf_(pq[SS+s], -6.0f, 4.0f);
        }
        float d = qm - pm;
        acc += plv - qlv + (__expf(qlv) + d*d) / (__expf(plv) + 1e-8f) - 1.0f;
    }
    red[threadIdx.x] = acc; __syncthreads();
    for (int o = 128; o > 0; o >>= 1) { if (threadIdx.x < o) red[threadIdx.x] += red[threadIdx.x+o]; __syncthreads(); }
    if (threadIdx.x == 0) partial[t] = red[0];
}

__global__ __launch_bounds__(256) void kl_final_kernel(const float* __restrict__ partial,
                                                       float* __restrict__ out)
{
    __shared__ float red[256];
    float acc = 0.0f;
    for (int i = threadIdx.x; i < TT; i += 256) acc += partial[i];
    red[threadIdx.x] = acc; __syncthreads();
    for (int o = 128; o > 0; o >>= 1) { if (threadIdx.x < o) red[threadIdx.x] += red[threadIdx.x+o]; __syncthreads(); }
    if (threadIdx.x == 0) out[0] = red[0] * 0.5f / ((float)TT * (float)BB);
}

__global__ void softmax4_kernel(const float* __restrict__ logits, float* __restrict__ probs)
{
    int r = blockIdx.x * blockDim.x + threadIdx.x;
    if (r >= BB*NSTEPS) return;
    const float* l = logits + (size_t)r*RR;
    float mx = fmaxf(fmaxf(l[0], l[1]), fmaxf(l[2], l[3]));
    float e0 = __expf(l[0]-mx), e1 = __expf(l[1]-mx), e2 = __expf(l[2]-mx), e3 = __expf(l[3]-mx);
    float inv = 1.0f / (e0+e1+e2+e3);
    float* p = probs + (size_t)r*RR;
    p[0]=e0*inv; p[1]=e1*inv; p[2]=e2*inv; p[3]=e3*inv;
}

// ---------------------------------------------------------------------------
// Host-side helpers
// ---------------------------------------------------------------------------
static void pack_frag(const float* W, __bf16* out, int Nreal, int K, int NtilesPad, hipStream_t s)
{
    int Kt = K / 32;
    size_t total = (size_t)Kt * NtilesPad * 512;
    int blocks = (int)((total + 255) / 256);
    pack_bfrag_kernel<<<blocks, 256, 0, s>>>(W, out, Nreal, K, NtilesPad);
}

static void gemm(const float* A, long lda, const __bf16* Bf, const float* bias,
                 float* C, long ldc, int Mt, int NtPad, int Kt, int Nvalid, int act, hipStream_t s)
{
    int Ng = NtPad >> 2;
    long tiles = (long)Mt * Ng;
    int blocks = (int)((tiles + 7) / 8);
    gemm_bf16_wmma<<<blocks, 256, 0, s>>>(A, lda, Bf, bias, C, ldc, Mt, NtPad, Kt, Nvalid, act);
}

extern "C" void kernel_launch(void* const* d_in, const int* in_sizes, int n_in,
                              void* d_out, int out_size, void* d_ws, size_t ws_size,
                              hipStream_t stream)
{
    (void)in_sizes; (void)n_in; (void)out_size; (void)ws_size;

    const float* x          = (const float*)d_in[0];
    const float* eps        = (const float*)d_in[1];
    const float* enc_Wih_f  = (const float*)d_in[2];
    const float* enc_Whh_f  = (const float*)d_in[3];
    const float* enc_bih_f  = (const float*)d_in[4];
    const float* enc_bhh_f  = (const float*)d_in[5];
    const float* enc_Wih_b  = (const float*)d_in[6];
    const float* enc_Whh_b  = (const float*)d_in[7];
    const float* enc_bih_b  = (const float*)d_in[8];
    const float* enc_bhh_b  = (const float*)d_in[9];
    const float* post_W     = (const float*)d_in[10];
    const float* post_b     = (const float*)d_in[11];
    const float* cell_Wih   = (const float*)d_in[12];
    const float* cell_Whh   = (const float*)d_in[13];
    const float* cell_bih   = (const float*)d_in[14];
    const float* cell_bhh   = (const float*)d_in[15];
    const float* prior_W    = (const float*)d_in[16];
    const float* prior_b    = (const float*)d_in[17];
    const float* em_ln_g    = (const float*)d_in[18];
    const float* em_ln_b    = (const float*)d_in[19];
    const float* em_W1      = (const float*)d_in[20];
    const float* em_b1      = (const float*)d_in[21];
    const float* em_W2      = (const float*)d_in[22];
    const float* em_b2      = (const float*)d_in[23];
    const float* rg_ln_g    = (const float*)d_in[24];
    const float* rg_ln_b    = (const float*)d_in[25];
    const float* rg_W       = (const float*)d_in[26];
    const float* rg_bias    = (const float*)d_in[27];
    // d_in[28] = steps (==4, constant in setup)

    float* ws  = (float*)d_ws;
    float* out = (float*)d_out;

    // ---------------- workspace layout (floats, all 16-float aligned) -------
    const size_t SZ_GX  = (size_t)BB*TT*G3;          // 50,331,648
    const size_t SZ_ENC = (size_t)TT*BB*2*SS;        // 33,554,432
    const size_t SZ_Z   = (size_t)TT*BB*SS;          // 16,777,216
    const size_t SZ_HS  = (size_t)(TT-1)*BB*SS;      // 16,744,448

    const size_t OFF_GXF  = 0;
    const size_t OFF_GXB  = OFF_GXF + SZ_GX;
    const size_t OFF_ENC  = OFF_GXB + SZ_GX;         // later reused for PR
    const size_t OFF_Z    = OFF_ENC + SZ_ENC;
    const size_t OFF_HS   = OFF_Z + SZ_Z;
    const size_t OFF_HL   = OFF_HS + SZ_HS;
    const size_t OFF_G    = OFF_HL + (size_t)BB*SS;
    const size_t OFF_GH   = OFF_G  + (size_t)BB*G3;
    const size_t OFF_PH   = OFF_GH + (size_t)BB*G3;
    const size_t OFF_ST   = OFF_PH + (size_t)BB*SS;
    const size_t OFF_PRR  = OFF_ST + (size_t)BB*SS;
    const size_t OFF_FS   = OFF_PRR + (size_t)BB*2*SS;
    const size_t OFF_LN1  = OFF_FS  + (size_t)BB*NSTEPS*SS;
    const size_t OFF_LN2  = OFF_LN1 + (size_t)BB*NSTEPS*SS;
    const size_t OFF_H2   = OFF_LN2 + (size_t)BB*NSTEPS*SS;
    const size_t OFF_KLP  = OFF_H2  + (size_t)BB*NSTEPS*SS;
    const size_t OFF_FRAG = (OFF_KLP + TT + 15) & ~(size_t)15;

    float* GXF = ws + OFF_GXF;          // gx forward; later reused as GZ
    float* GXB = ws + OFF_GXB;          // gx backward; later reused as PP
    float* ENC = ws + OFF_ENC;          // enc; later reused as PR
    float* Z   = ws + OFF_Z;
    float* HS  = ws + OFF_HS;
    float* HL  = ws + OFF_HL;
    float* GB  = ws + OFF_G;
    float* GHB = ws + OFF_GH;
    float* PH  = ws + OFF_PH;
    float* STB = ws + OFF_ST;
    float* PRR = ws + OFF_PRR;
    float* FS  = ws + OFF_FS;
    float* LN1 = ws + OFF_LN1;
    float* LN2 = ws + OFF_LN2;
    float* H2  = ws + OFF_H2;
    float* KLP = ws + OFF_KLP;
    float* GZ  = GXF;                   // aliases (sequenced safely)
    float* PP  = GXB;
    float* PR  = ENC;

    // Fragment region (bf16). All Ntiles padded to multiples of 4.
    __bf16* frag = (__bf16*)(ws + OFF_FRAG);
    const size_t FW_IHF   = 0;        // 4*48*512  =  98304
    const size_t FW_IHB   = 98304;    //              98304
    const size_t FW_HHF   = 196608;   // 8*48*512  = 196608
    const size_t FW_HHB   = 393216;   //             196608
    const size_t FW_CIH   = 589824;   //             196608
    const size_t FW_CHH   = 786432;   //             196608
    const size_t FW_POST  = 983040;   // 16*32*512 = 262144
    const size_t FW_PRIOR = 1245184;  // 8*32*512  = 131072
    const size_t FW_EMW1  = 1376256;  // 8*16*512  =  65536
    const size_t FW_EMW2  = 1441792;  // 8*4*512   =  16384 (2 tiles padded to 4)
    const size_t FW_RGW   = 1458176;  // 8*4*512   =  16384 (1 tile padded to 4)

    // d_out layout: obs[16384] | probs[2048] | logits[2048] | kl[1]
    const size_t OUT_OBS = 0, OUT_PROBS = 16384, OUT_LOGITS = 18432, OUT_KL = 20480;

    // ---------------- 1. pack all weights into WMMA bf16 fragments ---------
    pack_frag(enc_Wih_f, frag+FW_IHF,   G3,   DD,   48, stream);
    pack_frag(enc_Wih_b, frag+FW_IHB,   G3,   DD,   48, stream);
    pack_frag(enc_Whh_f, frag+FW_HHF,   G3,   SS,   48, stream);
    pack_frag(enc_Whh_b, frag+FW_HHB,   G3,   SS,   48, stream);
    pack_frag(cell_Wih,  frag+FW_CIH,   G3,   SS,   48, stream);
    pack_frag(cell_Whh,  frag+FW_CHH,   G3,   SS,   48, stream);
    pack_frag(post_W,    frag+FW_POST,  2*SS, 2*SS, 32, stream);
    pack_frag(prior_W,   frag+FW_PRIOR, 2*SS, SS,   32, stream);
    pack_frag(em_W1,     frag+FW_EMW1,  SS,   SS,   16, stream);
    pack_frag(em_W2,     frag+FW_EMW2,  OO,   SS,   4,  stream);
    pack_frag(rg_W,      frag+FW_RGW,   RR,   SS,   4,  stream);

    // ---------------- 2. input projections (big WMMA GEMMs) ----------------
    // gxf/gxb rows are (b*T + t); time reversal for backward handled in scan.
    gemm(x, DD, frag+FW_IHF, enc_bih_f, GXF, G3, (BB*TT)/16, 48, DD/32, G3, 0, stream);
    gemm(x, DD, frag+FW_IHB, enc_bih_b, GXB, G3, (BB*TT)/16, 48, DD/32, G3, 0, stream);

    // ---------------- 3. fwd + bwd encoder scans (2 WGPs, persistent) ------
    size_t scan_lds = (size_t)2 * BB * SS * sizeof(__bf16);   // 128 KB
    gru_scan_kernel<<<2, 1024, scan_lds, stream>>>(
        GXF, frag+FW_HHF, enc_bhh_f, ENC, 0,  2*SS, 0,
        GXB, frag+FW_HHB, enc_bhh_b, ENC, SS, 2*SS, 1,
        TT, /*tmajor=*/0, nullptr);

    // ---------------- 4. posterior projection + reparameterization ---------
    gemm(ENC, 2*SS, frag+FW_POST, post_b, PP, 2*SS, (TT*BB)/16, 32, (2*SS)/32, 2*SS, 0, stream);
    {
        size_t total = (size_t)TT*BB*SS;
        build_z_kernel<<<(int)((total + 255)/256), 256, 0, stream>>>(PP, eps, Z);
    }

    // ---------------- 5. cell input projection + cell scan -----------------
    gemm(Z, SS, frag+FW_CIH, cell_bih, GZ, G3, ((TT-1)*BB)/16, 48, SS/32, G3, 0, stream);
    gru_scan_kernel<<<1, 1024, scan_lds, stream>>>(
        GZ, frag+FW_CHH, cell_bhh, HS, 0, SS, 0,
        GZ, frag+FW_CHH, cell_bhh, HS, 0, SS, 0,
        TT-1, /*tmajor=*/1, HL);

    // ---------------- 6. prior projection + KL -----------------------------
    gemm(HS, SS, frag+FW_PRIOR, prior_b, PR, 2*SS, ((TT-1)*BB)/16, 32, SS/32, 2*SS, 0, stream);
    kl_partial_kernel<<<TT, 256, 0, stream>>>(PP, PR, KLP);
    kl_final_kernel<<<1, 256, 0, stream>>>(KLP, out + OUT_KL);

    // ---------------- 7. autoregressive rollout (4 steps) ------------------
    const float* st_src = Z + (size_t)(TT-1)*BB*SS;   // z[-1]
    const float* ph_src = HL;                         // h_last
    for (int step = 0; step < NSTEPS; ++step) {
        gemm(st_src, SS, frag+FW_CIH, cell_bih, GB,  G3, BB/16, 48, SS/32, G3, 0, stream);
        gemm(ph_src, SS, frag+FW_CHH, cell_bhh, GHB, G3, BB/16, 48, SS/32, G3, 0, stream);
        gru_gate_kernel<<<(BB*SS + 255)/256, 256, 0, stream>>>(GB, GHB, ph_src, PH);
        gemm(PH, SS, frag+FW_PRIOR, prior_b, PRR, 2*SS, BB/16, 32, SS/32, 2*SS, 0, stream);
        extract_mean_kernel<<<(BB*SS + 255)/256, 256, 0, stream>>>(PRR, STB, FS, step);
        st_src = STB;
        ph_src = PH;
    }

    // ---------------- 8. heads ---------------------------------------------
    const int ROWS = BB * NSTEPS;  // 512
    ln_gelu_kernel<<<ROWS, 256, 0, stream>>>(FS, em_ln_g, em_ln_b, LN1, ROWS);
    ln_gelu_kernel<<<ROWS, 256, 0, stream>>>(FS, rg_ln_g, rg_ln_b, LN2, ROWS);
    gemm(LN1, SS, frag+FW_EMW1, em_b1, H2, SS, ROWS/16, 16, SS/32, SS, /*GELU*/1, stream);
    gemm(H2,  SS, frag+FW_EMW2, em_b2, out + OUT_OBS,    OO, ROWS/16, 4, SS/32, OO, 0, stream);
    gemm(LN2, SS, frag+FW_RGW,  rg_bias, out + OUT_LOGITS, RR, ROWS/16, 4, SS/32, RR, 0, stream);
    softmax4_kernel<<<(ROWS + 255)/256, 256, 0, stream>>>(out + OUT_LOGITS, out + OUT_PROBS);
}
